// Model_53008486367722
// MI455X (gfx1250) — compile-verified, matrix operand order checked by reading, not run on hardware
//
#include <hip/hip_runtime.h>
#include <math.h>

// ---------------- model dims ----------------
#define BN      32
#define CN      21
#define LSEQ    512
#define PREDL   96
#define DMODEL  512
#define NHEAD   8
#define EHEAD   64
#define DFFN    2048
#define NLAY    3
#define PNUM    64
#define NBC     (BN*CN)        /* 672  */
#define MROWS   (NBC*PNUM)     /* 43008 */
#define TOPK    4
#define MAHALF  12             /* (25-1)/2 */

typedef unsigned short u16;
typedef unsigned int   u32;

// ---------------- WMMA types ----------------
typedef __attribute__((ext_vector_type(16))) __bf16 v16bf;
typedef __attribute__((ext_vector_type(8)))  float  v8f;

struct __align__(16) U4 { u32 x, y, z, w; };
union  FragU { v16bf v; U4 q[2]; };

__device__ __forceinline__ u16 f2bf(float f) {
    u32 u = __float_as_uint(f);
    u32 r = (u + 0x7FFFu + ((u >> 16) & 1u)) >> 16;   // RNE
    return (u16)r;
}

// =====================================================================
// WMMA GEMM:  C[M,N] = A[M,K](bf16,row-major) * W^T[N,K](bf16) + bias
// block = 256 threads = 8 waves (4M x 2N); each wave: 16(M) x 64(N)
// -> one A fragment reused against 4 B fragments per K-step (4 WMMA).
// Block tile: 64(M) x 128(N).
// GUARD=0: M%64==0 and N%128==0 assumed -> branch-free inner loop.
// All tile coordinates are forced into SGPRs via readfirstlane so any
// remaining guards are scalar branches (EXEC stays all-1s for WMMA).
// =====================================================================
template<int GUARD>
__global__ void gemm_wmma(const u16* __restrict__ A, const u16* __restrict__ BT,
                          const float* __restrict__ bias,
                          float* __restrict__ Cf, u16* __restrict__ Cb,
                          int M, int N, int K, int relu)
{
    const int lane = threadIdx.x & 31;
    const int wave = __builtin_amdgcn_readfirstlane((int)(threadIdx.x >> 5)); // SGPR
    const int mw = wave & 3, nw = wave >> 2;
    const int M0 = blockIdx.x * 64 + mw * 16;        // scalar
    const int Ng = blockIdx.y * 128 + nw * 64;       // scalar
    if (GUARD) { if (M0 >= M || Ng >= N) return; }   // scalar branch

    bool v0 = true, v1 = true, v2 = true, v3 = true;
    if (GUARD) {
        v1 = (Ng + 16) < N;
        v2 = (Ng + 32) < N;
        v3 = (Ng + 48) < N;
    }

    v8f acc[4];
    #pragma unroll
    for (int j = 0; j < 4; ++j) acc[j] = (v8f){0.f,0.f,0.f,0.f,0.f,0.f,0.f,0.f};

    const int r = lane & 15;                // row within tile
    const int h = lane >> 4;                // K-half select
    const u16* pa  = A  + (size_t)(M0 + r) * K + (h << 3);
    const u16* pb0 = BT + (size_t)(Ng +  0 + r) * K + (h << 3);
    const u16* pb1 = BT + (size_t)(Ng + 16 + r) * K + (h << 3);
    const u16* pb2 = BT + (size_t)(Ng + 32 + r) * K + (h << 3);
    const u16* pb3 = BT + (size_t)(Ng + 48 + r) * K + (h << 3);

    #pragma unroll 2
    for (int k = 0; k < K; k += 32) {
        FragU fa, fb0, fb1, fb2, fb3;
        fa.q[0]  = *(const U4*)(pa  + k);        // K = k + h*8 .. +7
        fa.q[1]  = *(const U4*)(pa  + k + 16);   // K = k + 16 + h*8 ..
        fb0.q[0] = *(const U4*)(pb0 + k);
        fb0.q[1] = *(const U4*)(pb0 + k + 16);
        if (!GUARD || v1) { fb1.q[0] = *(const U4*)(pb1 + k);
                            fb1.q[1] = *(const U4*)(pb1 + k + 16); }
        if (!GUARD || v2) { fb2.q[0] = *(const U4*)(pb2 + k);
                            fb2.q[1] = *(const U4*)(pb2 + k + 16); }
        if (!GUARD || v3) { fb3.q[0] = *(const U4*)(pb3 + k);
                            fb3.q[1] = *(const U4*)(pb3 + k + 16); }
        if (k + 512 < K) __builtin_prefetch((const void*)(pa + k + 512), 0, 3);

        acc[0] = __builtin_amdgcn_wmma_f32_16x16x32_bf16(
                     false, fa.v, false, fb0.v, (short)0, acc[0], false, false);
        if (!GUARD || v1)
            acc[1] = __builtin_amdgcn_wmma_f32_16x16x32_bf16(
                         false, fa.v, false, fb1.v, (short)0, acc[1], false, false);
        if (!GUARD || v2)
            acc[2] = __builtin_amdgcn_wmma_f32_16x16x32_bf16(
                         false, fa.v, false, fb2.v, (short)0, acc[2], false, false);
        if (!GUARD || v3)
            acc[3] = __builtin_amdgcn_wmma_f32_16x16x32_bf16(
                         false, fa.v, false, fb3.v, (short)0, acc[3], false, false);
    }

    // C/D layout: lane l -> n = l&15 ; VGPR i -> m = i + 8*(l>>4)
    const int n  = lane & 15;
    const int mh = (lane >> 4) * 8;
    #pragma unroll
    for (int j = 0; j < 4; ++j) {
        if (GUARD) {
            if (j == 1 && !v1) continue;
            if (j == 2 && !v2) continue;
            if (j == 3 && !v3) continue;
        }
        const int N0 = Ng + j * 16;
        const float bv = bias ? bias[N0 + n] : 0.f;
        #pragma unroll
        for (int i = 0; i < 8; ++i) {
            float v = acc[j][i] + bv;
            if (relu) v = fmaxf(v, 0.f);
            size_t off = (size_t)(M0 + mh + i) * N + N0 + n;
            if (Cf) Cf[off] = v;
            if (Cb) Cb[off] = f2bf(v);
        }
    }
}

// =====================================================================
// Weight transpose + f32 -> bf16 :  WT[n*K+k] = bf16(W[k*N+n])
// =====================================================================
__global__ void conv_wT(const float* __restrict__ W, u16* __restrict__ WT,
                        int K, int N, unsigned tot)
{
    unsigned i = blockIdx.x * 256u + threadIdx.x;
    if (i >= tot) return;
    unsigned n = i / (unsigned)K, k = i % (unsigned)K;
    WT[i] = f2bf(W[(size_t)k * N + n]);
}

// =====================================================================
// Instance-norm statistics per (b,c) over L_SEQ
// =====================================================================
__global__ void inorm_stats(const float* __restrict__ x, float* __restrict__ mean,
                            float* __restrict__ stdv)
{
    int bc = blockIdx.x, b = bc / CN, c = bc % CN;
    float s = 0.f, s2 = 0.f;
    for (int t = threadIdx.x; t < LSEQ; t += 256) {
        float v = x[((size_t)b * LSEQ + t) * CN + c];
        s += v; s2 += v * v;
    }
    __shared__ float r1[256], r2[256];
    r1[threadIdx.x] = s; r2[threadIdx.x] = s2; __syncthreads();
    for (int o = 128; o > 0; o >>= 1) {
        if (threadIdx.x < o) { r1[threadIdx.x] += r1[threadIdx.x+o];
                               r2[threadIdx.x] += r2[threadIdx.x+o]; }
        __syncthreads();
    }
    if (threadIdx.x == 0) {
        float m = r1[0] / (float)LSEQ;
        float var = r2[0] / (float)LSEQ - m * m;
        mean[bc] = m; stdv[bc] = sqrtf(var + 1e-5f);
    }
}

// =====================================================================
// Patch embedding + positional embedding : x[NBC, PNUM, D]
// =====================================================================
__global__ void patch_embed(const float* __restrict__ x_enc,
                            const float* __restrict__ mean, const float* __restrict__ stdv,
                            const float* __restrict__ w_val,
                            float* __restrict__ xF, u16* __restrict__ xB)
{
    int n = blockIdx.x, p = blockIdx.y;
    int b = n / CN, c = n % CN;
    __shared__ float patch[16];
    if (threadIdx.x < 16) {
        int t = p * 8 + threadIdx.x;             // stride 8, replicate-pad
        if (t > LSEQ - 1) t = LSEQ - 1;
        patch[threadIdx.x] = (x_enc[((size_t)b * LSEQ + t) * CN + c] - mean[n]) / stdv[n];
    }
    __syncthreads();
    for (int d = threadIdx.x; d < DMODEL; d += 256) {
        float acc = 0.f;
        #pragma unroll
        for (int j = 0; j < 16; ++j) acc += patch[j] * w_val[j * DMODEL + d];
        float freq = expf(-(float)(d & ~1) * (logf(10000.f) / (float)DMODEL));
        float ang  = (float)p * freq;
        acc += (d & 1) ? cosf(ang) : sinf(ang);
        size_t idx = ((size_t)n * PNUM + p) * DMODEL + d;
        xF[idx] = acc; xB[idx] = f2bf(acc);
    }
}

// =====================================================================
// mean_corr[n,tau] = (1/D) sum_{t,d} q[n,(t+tau)%L,d] * k[n,t,d]
// (analytic equivalent of irfft(rfft(q) * conj(rfft(k))) mean over H,E)
// =====================================================================
__global__ void mean_corr_kernel(const float* __restrict__ q, const float* __restrict__ k,
                                 float* __restrict__ mc)
{
    int n = blockIdx.x, tau = threadIdx.x;       // 64 threads
    __shared__ float qs[64][128];
    __shared__ float ks[64][128];
    float acc = 0.f;
    for (int c = 0; c < DMODEL; c += 128) {
        for (int i = threadIdx.x; i < 64 * 128; i += 64) {
            int t = i >> 7, d = i & 127;
            size_t g = ((size_t)n * PNUM + t) * DMODEL + c + d;
            qs[t][d] = q[g]; ks[t][d] = k[g];
        }
        __syncthreads();
        for (int t = 0; t < 64; ++t) {
            int tt = (t + tau) & 63;
            #pragma unroll 8
            for (int d = 0; d < 128; ++d) acc += qs[tt][d] * ks[t][d];
        }
        __syncthreads();
    }
    mc[n * PNUM + tau] = acc * (1.f / (float)DMODEL);
}

// =====================================================================
// batch-mean over n, then top-4 delays
// =====================================================================
__global__ void topk_kernel(const float* __restrict__ mc, int* __restrict__ topidx)
{
    int tau = threadIdx.x;                       // 64 threads
    float s = 0.f;
    for (int n = 0; n < NBC; ++n) s += mc[n * PNUM + tau];
    __shared__ float gs[64];
    gs[tau] = s; __syncthreads();
    if (tau == 0) {
        bool used[64];
        for (int t = 0; t < 64; ++t) used[t] = false;
        for (int kk = 0; kk < TOPK; ++kk) {
            float best = -1e30f; int bi = 0;
            for (int t = 0; t < 64; ++t)
                if (!used[t] && gs[t] > best) { best = gs[t]; bi = t; }
            used[bi] = true; topidx[kk] = bi;
        }
    }
}

// =====================================================================
// time-delay aggregation: softmax over top-k corr, roll v
// =====================================================================
__global__ void agg_kernel(const float* __restrict__ mc, const int* __restrict__ topidx,
                           const float* __restrict__ v,
                           float* __restrict__ outF, u16* __restrict__ outB)
{
    int n = blockIdx.x;
    int   id[TOPK]; float w[TOPK];
    float mx = -1e30f;
    #pragma unroll
    for (int kk = 0; kk < TOPK; ++kk) {
        id[kk] = topidx[kk];
        w[kk]  = mc[n * PNUM + id[kk]];
        mx = fmaxf(mx, w[kk]);
    }
    float sum = 0.f;
    #pragma unroll
    for (int kk = 0; kk < TOPK; ++kk) { w[kk] = expf(w[kk] - mx); sum += w[kk]; }
    float inv = 1.f / sum;
    #pragma unroll
    for (int kk = 0; kk < TOPK; ++kk) w[kk] *= inv;

    for (int i = threadIdx.x; i < PNUM * DMODEL; i += 256) {
        int t = i >> 9, d = i & (DMODEL - 1);
        float acc = 0.f;
        #pragma unroll
        for (int kk = 0; kk < TOPK; ++kk) {
            int ts = (t + id[kk]) & 63;
            acc += w[kk] * v[((size_t)n * PNUM + ts) * DMODEL + d];
        }
        size_t o = ((size_t)n * PNUM + t) * DMODEL + d;
        outF[o] = acc; outB[o] = f2bf(acc);
    }
}

// =====================================================================
// full softmax attention, per (n, head); L=64, E=64
// =====================================================================
__global__ void full_attn_kernel(const float* __restrict__ q, const float* __restrict__ k,
                                 const float* __restrict__ v,
                                 float* __restrict__ outF, u16* __restrict__ outB)
{
    int nh = blockIdx.x, n = nh / NHEAD, h = nh % NHEAD;
    int l = threadIdx.x;                          // 64 threads
    __shared__ float ks[64][EHEAD + 1];
    __shared__ float vs[64][EHEAD + 1];
    __shared__ float ss[64][64 + 1];

    size_t base = ((size_t)n * PNUM + l) * DMODEL + h * EHEAD;
    float qr[EHEAD];
    #pragma unroll
    for (int e = 0; e < EHEAD; ++e) {
        qr[e]    = q[base + e];
        ks[l][e] = k[base + e];
        vs[l][e] = v[base + e];
    }
    __syncthreads();

    for (int s = 0; s < 64; ++s) {
        float d = 0.f;
        #pragma unroll
        for (int e = 0; e < EHEAD; ++e) d += qr[e] * ks[s][e];
        ss[l][s] = d * 0.125f;                    // 1/sqrt(64)
    }
    float mx = -1e30f;
    for (int s = 0; s < 64; ++s) mx = fmaxf(mx, ss[l][s]);
    float sum = 0.f;
    for (int s = 0; s < 64; ++s) { float e_ = expf(ss[l][s] - mx); ss[l][s] = e_; sum += e_; }
    float inv = 1.f / sum;

    float acc[EHEAD];
    #pragma unroll
    for (int e = 0; e < EHEAD; ++e) acc[e] = 0.f;
    for (int s = 0; s < 64; ++s) {
        float p = ss[l][s] * inv;
        #pragma unroll
        for (int e = 0; e < EHEAD; ++e) acc[e] += p * vs[s][e];
    }
    #pragma unroll
    for (int e = 0; e < EHEAD; ++e) { outF[base + e] = acc[e]; outB[base + e] = f2bf(acc[e]); }
}

// =====================================================================
// concat [a | s] -> bf16 [M, 2D]
// =====================================================================
__global__ void concat_kernel(const float* __restrict__ a, const float* __restrict__ s,
                              u16* __restrict__ cat)
{
    size_t i = (size_t)blockIdx.x * 256 + threadIdx.x;
    if (i >= (size_t)MROWS * 2 * DMODEL) return;
    size_t m = i / (2 * DMODEL);
    int    j = (int)(i % (2 * DMODEL));
    float  v = (j < DMODEL) ? a[m * DMODEL + j] : s[m * DMODEL + (j - DMODEL)];
    cat[i] = f2bf(v);
}

// =====================================================================
// x = series_decomp(x + f): residual after MA-25 with edge replication
// block = one sequence n, 512 threads (d); running-window sum fully
// unrolled so all register-array indices constant-fold.
// =====================================================================
__global__ void add_decomp_kernel(const float* __restrict__ x, const float* __restrict__ f,
                                  float* __restrict__ xo, u16* __restrict__ xb)
{
    int n = blockIdx.x, d = threadIdx.x;
    float s[PNUM];
    #pragma unroll
    for (int t = 0; t < PNUM; ++t) {
        size_t idx = ((size_t)n * PNUM + t) * DMODEL + d;
        s[t] = x[idx] + f[idx];
    }
    float win = 12.f * s[0];
    #pragma unroll
    for (int j = 0; j <= MAHALF; ++j) win += s[j];
    #pragma unroll
    for (int t = 0; t < PNUM; ++t) {
        float r = s[t] - win * (1.f / 25.f);
        size_t idx = ((size_t)n * PNUM + t) * DMODEL + d;
        xo[idx] = r; xb[idx] = f2bf(r);
        int nxt = (t + MAHALF + 1 > PNUM - 1) ? PNUM - 1 : t + MAHALF + 1;
        int prv = (t - MAHALF < 0) ? 0 : t - MAHALF;
        win += s[nxt] - s[prv];
    }
}

// =====================================================================
// batchnorm stats over all rows, per channel d
// =====================================================================
__global__ void bn_stats(const float* __restrict__ x, float* __restrict__ sums)
{
    int d = blockIdx.x;
    float s = 0.f, s2 = 0.f;
    for (int m = threadIdx.x; m < MROWS; m += 256) {
        float v = x[(size_t)m * DMODEL + d];
        s += v; s2 += v * v;
    }
    __shared__ float r1[256], r2[256];
    r1[threadIdx.x] = s; r2[threadIdx.x] = s2; __syncthreads();
    for (int o = 128; o > 0; o >>= 1) {
        if (threadIdx.x < o) { r1[threadIdx.x] += r1[threadIdx.x+o];
                               r2[threadIdx.x] += r2[threadIdx.x+o]; }
        __syncthreads();
    }
    if (threadIdx.x == 0) { sums[d] = r1[0]; sums[DMODEL + d] = r2[0]; }
}

// =====================================================================
// batchnorm apply + transpose (P,D)->(D,P) into bf16 head input
// =====================================================================
__global__ void bn_apply_t(const float* __restrict__ x, const float* __restrict__ sums,
                           const float* __restrict__ g, const float* __restrict__ bb,
                           u16* __restrict__ enc)
{
    int n = blockIdx.x;
    for (int i = threadIdx.x; i < DMODEL * PNUM; i += 256) {
        int d = i >> 6, p = i & 63;
        float mu  = sums[d] * (1.f / (float)MROWS);
        float var = sums[DMODEL + d] * (1.f / (float)MROWS) - mu * mu;
        float inv = rsqrtf(var + 1e-5f);
        float v = (x[((size_t)n * PNUM + p) * DMODEL + d] - mu) * inv * g[d] + bb[d];
        enc[(size_t)n * (DMODEL * PNUM) + i] = f2bf(v);
    }
}

// =====================================================================
// de-normalize + transpose to [B, PRED, C]
// =====================================================================
__global__ void head_out(const float* __restrict__ dec, const float* __restrict__ mean,
                         const float* __restrict__ stdv, float* __restrict__ out)
{
    int i = blockIdx.x * 256 + threadIdx.x;
    if (i >= BN * PREDL * CN) return;
    int c = i % CN, p = (i / CN) % PREDL, b = i / (CN * PREDL);
    int n = b * CN + c;
    out[i] = dec[(size_t)n * PREDL + p] * stdv[n] + mean[n];
}

// =====================================================================
// host: orchestration
// =====================================================================
extern "C" void kernel_launch(void* const* d_in, const int* in_sizes, int n_in,
                              void* d_out, int out_size, void* d_ws, size_t ws_size,
                              hipStream_t stream)
{
    (void)in_sizes; (void)n_in; (void)out_size; (void)ws_size;

    const float* x_enc = (const float*)d_in[0];
    const float* w_val = (const float*)d_in[1];
    const float* Wq_ac = (const float*)d_in[2];
    const float* Wk_ac = (const float*)d_in[3];
    const float* Wv_ac = (const float*)d_in[4];
    const float* Wo_ac = (const float*)d_in[5];
    const float* bq_ac = (const float*)d_in[6];
    const float* bk_ac = (const float*)d_in[7];
    const float* bv_ac = (const float*)d_in[8];
    const float* bo_ac = (const float*)d_in[9];
    const float* Wq_sa = (const float*)d_in[10];
    const float* Wk_sa = (const float*)d_in[11];
    const float* Wv_sa = (const float*)d_in[12];
    const float* Wo_sa = (const float*)d_in[13];
    const float* bq_sa = (const float*)d_in[14];
    const float* bk_sa = (const float*)d_in[15];
    const float* bv_sa = (const float*)d_in[16];
    const float* bo_sa = (const float*)d_in[17];
    const float* Wf    = (const float*)d_in[18];
    const float* bff   = (const float*)d_in[19];
    const float* W1    = (const float*)d_in[20];
    const float* W2    = (const float*)d_in[21];
    const float* bn_g  = (const float*)d_in[22];
    const float* bn_b  = (const float*)d_in[23];
    const float* Whead = (const float*)d_in[24];
    const float* bhead = (const float*)d_in[25];

    const size_t NLD = (size_t)MROWS * DMODEL;

    char* wsb = (char*)d_ws;
    size_t off = 0;
    auto alloc = [&](size_t bytes) -> void* {
        void* p = wsb + off;
        off += (bytes + 255) & ~(size_t)255;
        return p;
    };

    float* meanB = (float*)alloc(NBC * 4);
    float* stdB  = (float*)alloc(NBC * 4);
    float* xF    = (float*)alloc(NLD * 4);
    u16*   xH    = (u16*)  alloc(NLD * 2);
    float* qF    = (float*)alloc(NLD * 4);
    float* kF    = (float*)alloc(NLD * 4);
    float* vF    = (float*)alloc(NLD * 4);
    float* t1F   = (float*)alloc(NLD * 4);
    u16*   t1H   = (u16*)  alloc(NLD * 2);
    float* aF    = (float*)alloc(NLD * 4);
    u16*   catH  = (u16*)  alloc(NLD * 2 * 2);        // [M, 2D]
    u16*   y1H   = (u16*)  alloc((size_t)MROWS * DFFN * 2);
    float* mc    = (float*)alloc((size_t)NBC * PNUM * 4);
    int*   topi  = (int*)  alloc(64);
    float* bnsum = (float*)alloc(2 * DMODEL * 4);
    u16*   encH  = (u16*)  alloc(NLD * 2);            // [NBC, D*P]
    float* decF  = (float*)alloc((size_t)NBC * PREDL * 4);

    const size_t DD = (size_t)DMODEL * DMODEL;
    u16 *WqacT[NLAY], *WkacT[NLAY], *WvacT[NLAY], *WoacT[NLAY];
    u16 *WqsaT[NLAY], *WksaT[NLAY], *WvsaT[NLAY], *WosaT[NLAY];
    u16 *WfT[NLAY], *W1T[NLAY], *W2T[NLAY];
    for (int l = 0; l < NLAY; ++l) {
        WqacT[l] = (u16*)alloc(DD * 2); WkacT[l] = (u16*)alloc(DD * 2);
        WvacT[l] = (u16*)alloc(DD * 2); WoacT[l] = (u16*)alloc(DD * 2);
        WqsaT[l] = (u16*)alloc(DD * 2); WksaT[l] = (u16*)alloc(DD * 2);
        WvsaT[l] = (u16*)alloc(DD * 2); WosaT[l] = (u16*)alloc(DD * 2);
        WfT[l] = (u16*)alloc((size_t)2 * DMODEL * DMODEL * 2);
        W1T[l] = (u16*)alloc((size_t)DMODEL * DFFN * 2);
        W2T[l] = (u16*)alloc((size_t)DFFN * DMODEL * 2);
    }
    u16* WheadT = (u16*)alloc((size_t)DMODEL * PNUM * PREDL * 2);

    auto convT = [&](const float* W, u16* WT, int K, int N) {
        unsigned tot = (unsigned)((size_t)K * N);
        conv_wT<<<(tot + 255) / 256, 256, 0, stream>>>(W, WT, K, N, tot);
    };
    // interior GEMMs: M%64==0, N%128==0 -> branch-free kernel
    auto gemm = [&](const u16* A, const u16* WT, const float* bias,
                    float* Cf, u16* Cb, int M, int N, int K, int relu) {
        dim3 g(M / 64, N / 128);
        gemm_wmma<0><<<g, 256, 0, stream>>>(A, WT, bias, Cf, Cb, M, N, K, relu);
    };
    auto gemm_g = [&](const u16* A, const u16* WT, const float* bias,
                      float* Cf, u16* Cb, int M, int N, int K, int relu) {
        dim3 g((M + 63) / 64, (N + 127) / 128);
        gemm_wmma<1><<<g, 256, 0, stream>>>(A, WT, bias, Cf, Cb, M, N, K, relu);
    };

    // ---- weight conversion (transpose to [N,K] bf16) ----
    for (int l = 0; l < NLAY; ++l) {
        convT(Wq_ac + l * DD, WqacT[l], DMODEL, DMODEL);
        convT(Wk_ac + l * DD, WkacT[l], DMODEL, DMODEL);
        convT(Wv_ac + l * DD, WvacT[l], DMODEL, DMODEL);
        convT(Wo_ac + l * DD, WoacT[l], DMODEL, DMODEL);
        convT(Wq_sa + l * DD, WqsaT[l], DMODEL, DMODEL);
        convT(Wk_sa + l * DD, WksaT[l], DMODEL, DMODEL);
        convT(Wv_sa + l * DD, WvsaT[l], DMODEL, DMODEL);
        convT(Wo_sa + l * DD, WosaT[l], DMODEL, DMODEL);
        convT(Wf + (size_t)l * 2 * DMODEL * DMODEL, WfT[l], 2 * DMODEL, DMODEL);
        convT(W1 + (size_t)l * DMODEL * DFFN, W1T[l], DMODEL, DFFN);
        convT(W2 + (size_t)l * DFFN * DMODEL, W2T[l], DFFN, DMODEL);
    }
    convT(Whead, WheadT, DMODEL * PNUM, PREDL);

    // ---- instance norm + patch embedding ----
    inorm_stats<<<NBC, 256, 0, stream>>>(x_enc, meanB, stdB);
    patch_embed<<<dim3(NBC, PNUM), 256, 0, stream>>>(x_enc, meanB, stdB, w_val, xF, xH);

    // ---- encoder layers ----
    for (int l = 0; l < NLAY; ++l) {
        // AutoCorrelation branch
        gemm(xH, WqacT[l], bq_ac + l * DMODEL, qF, nullptr, MROWS, DMODEL, DMODEL, 0);
        gemm(xH, WkacT[l], bk_ac + l * DMODEL, kF, nullptr, MROWS, DMODEL, DMODEL, 0);
        gemm(xH, WvacT[l], bv_ac + l * DMODEL, vF, nullptr, MROWS, DMODEL, DMODEL, 0);
        mean_corr_kernel<<<NBC, 64, 0, stream>>>(qF, kF, mc);
        topk_kernel<<<1, 64, 0, stream>>>(mc, topi);
        agg_kernel<<<NBC, 256, 0, stream>>>(mc, topi, vF, t1F, t1H);
        gemm(t1H, WoacT[l], bo_ac + l * DMODEL, aF, nullptr, MROWS, DMODEL, DMODEL, 0);

        // Full attention branch
        gemm(xH, WqsaT[l], bq_sa + l * DMODEL, qF, nullptr, MROWS, DMODEL, DMODEL, 0);
        gemm(xH, WksaT[l], bk_sa + l * DMODEL, kF, nullptr, MROWS, DMODEL, DMODEL, 0);
        gemm(xH, WvsaT[l], bv_sa + l * DMODEL, vF, nullptr, MROWS, DMODEL, DMODEL, 0);
        full_attn_kernel<<<NBC * NHEAD, 64, 0, stream>>>(qF, kF, vF, t1F, t1H);
        gemm(t1H, WosaT[l], bo_sa + l * DMODEL, qF /*=s*/, nullptr, MROWS, DMODEL, DMODEL, 0);

        // fuse + decomp
        size_t catElems = (size_t)MROWS * 2 * DMODEL;
        concat_kernel<<<(unsigned)((catElems + 255) / 256), 256, 0, stream>>>(aF, qF, catH);
        gemm(catH, WfT[l], bff + l * DMODEL, t1F, nullptr, MROWS, DMODEL, 2 * DMODEL, 0);
        add_decomp_kernel<<<NBC, DMODEL, 0, stream>>>(xF, t1F, xF, xH);

        // FFN + decomp
        gemm(xH, W1T[l], nullptr, nullptr, y1H, MROWS, DFFN, DMODEL, 1);
        gemm(y1H, W2T[l], nullptr, t1F, nullptr, MROWS, DMODEL, DFFN, 0);
        add_decomp_kernel<<<NBC, DMODEL, 0, stream>>>(xF, t1F, xF, xH);
    }

    // ---- batchnorm + head ----
    bn_stats<<<DMODEL, 256, 0, stream>>>(xF, bnsum);
    bn_apply_t<<<NBC, 256, 0, stream>>>(xF, bnsum, bn_g, bn_b, encH);
    gemm_g(encH, WheadT, bhead, decF, nullptr, NBC, PREDL, DMODEL * PNUM, 0);
    head_out<<<(BN * PREDL * CN + 255) / 256, 256, 0, stream>>>(decF, meanB, stdB,
                                                                (float*)d_out);
}
